// SwinTransformerBlock_24584392802500
// MI455X (gfx1250) — matrix operand order repaired and checked by reading, not compiled, not run on hardware
//
#include <hip/hip_runtime.h>
#include <hip/hip_bf16.h>
#include <cstdint>
#include <cstddef>

typedef _Float16 v16h __attribute__((ext_vector_type(16)));
typedef _Float16 v8h  __attribute__((ext_vector_type(8)));
typedef float    v8f  __attribute__((ext_vector_type(8)));

__device__ __forceinline__ v16h mk16(v8h lo, v8h hi) {
  return __builtin_shufflevector(lo, hi, 0,1,2,3,4,5,6,7,8,9,10,11,12,13,14,15);
}

__device__ __forceinline__ v8f wmma_f16(v16h a, v16h b, v8f c) {
  return __builtin_amdgcn_wmma_f32_16x16x32_f16(false, a, false, b, (short)0, c,
                                                false, false);
}

// ---------------------------------------------------------------------------
// Problem constants: B=64, C=192, IMG=56, WS=7, SS=3, NH=6, N=49, dh=32,
// nW=64, B_=4096, tokens M=200704 (divisible by 256), Hn=768
// ---------------------------------------------------------------------------
#define M_TOK   200704
#define C_DIM   192
#define HN_DIM  768
#define NHEAD   6
#define DH      32

// ---------------------------------------------------------------------------
// Workspace layout (bytes, 256-aligned). FF aliases Xwin/Q/K (dead by then),
// H2 aliases Obuf (dead after proj GEMM).
// ---------------------------------------------------------------------------
#define XWIN_OFF   ((size_t)0)                    // 200704*192*2   = 77070336
#define Q_OFF      ((size_t)77070336)             // 24576*64*32*2  = 100663296
#define K_OFF      ((size_t)177733632)
#define VT_OFF     ((size_t)278396928)
#define OBUF_OFF   ((size_t)379060224)            // 77070336
#define H2_OFF     OBUF_OFF
#define XS_OFF     ((size_t)456130560)            // 200704*192*4   = 154140672
#define PBQKV_OFF  ((size_t)610271232)            // 576*192*2
#define PBPROJ_OFF (PBQKV_OFF + 221184)
#define PBFC1_OFF  (PBPROJ_OFF + 73728)
#define PBFC2_OFF  (PBFC1_OFF + 294912)
#define FF_OFF     ((size_t)0)                    // 200704*768*2   = 308281344 (aliases)

// ---------------------------------------------------------------------------
// Weight packing: f32 W[NC rows (out), K cols (in)]  ->  f16 fragments
// layout [kt][nt][lane][e] : n = nt*16 + (lane&15), k = kt*32 + (lane>>4)*16 + e
// (contiguous k order, same as the A fragments in gemm_wmma)
// ---------------------------------------------------------------------------
__global__ void pack_w(const float* __restrict__ w, _Float16* __restrict__ out,
                       int K, int NC) {
  int idx = blockIdx.x * 256 + threadIdx.x;
  if (idx >= K * NC) return;
  int e    = idx & 15;
  int lane = (idx >> 4) & 31;
  int nt   = (idx >> 9) % (NC / 16);
  int kt   = idx / (512 * (NC / 16));
  int n = nt * 16 + (lane & 15);
  int k = kt * 32 + (lane >> 4) * 16 + e;
  out[idx] = (_Float16)w[(size_t)n * K + k];
}

// ---------------------------------------------------------------------------
// LN0 fused with roll(-3,-3) + 7x7 window partition. Wave-per-token (wave32).
// Output token order: b_*49 + n, b_ = b*64 + wy*8 + wx, n = y*7 + x.
// Source pixel: r = (wy*7+y+3)%56, c = (wx*7+x+3)%56  (from NCHW input).
// ---------------------------------------------------------------------------
__global__ __launch_bounds__(256) void ln0_kernel(const float* __restrict__ x,
                                                  const float* __restrict__ g,
                                                  const float* __restrict__ be,
                                                  _Float16* __restrict__ Xwin) {
  const int wave = threadIdx.x >> 5, lane = threadIdx.x & 31;
  const int t = blockIdx.x * 8 + wave;
  const int b_ = t / 49, n = t % 49;
  const int b = b_ >> 6, wi = b_ & 63;
  const int r = ((wi >> 3) * 7 + n / 7 + 3) % 56;
  const int c = ((wi & 7) * 7 + n % 7 + 3) % 56;
  const float* xp = x + (size_t)b * C_DIM * 3136 + r * 56 + c;
  float v[6], sum = 0.f, sq = 0.f;
#pragma unroll
  for (int i = 0; i < 6; ++i) {
    v[i] = xp[(size_t)(lane + 32 * i) * 3136];
    sum += v[i]; sq += v[i] * v[i];
  }
#pragma unroll
  for (int m = 16; m >= 1; m >>= 1) {
    sum += __shfl_xor(sum, m, 32);
    sq  += __shfl_xor(sq, m, 32);
  }
  const float mu = sum * (1.f / C_DIM);
  const float rstd = rsqrtf(sq * (1.f / C_DIM) - mu * mu + 1e-5f);
  _Float16* op = Xwin + (size_t)t * C_DIM;
#pragma unroll
  for (int i = 0; i < 6; ++i) {
    int ch = lane + 32 * i;
    op[ch] = (_Float16)((v[i] - mu) * rstd * g[ch] + be[ch]);
  }
}

// LN1: input XS is flat NHWC f32 [M_TOK,192] contiguous per token.
__global__ __launch_bounds__(256) void ln1_kernel(const float* __restrict__ XS,
                                                  const float* __restrict__ g,
                                                  const float* __restrict__ be,
                                                  _Float16* __restrict__ H2) {
  const int wave = threadIdx.x >> 5, lane = threadIdx.x & 31;
  const int t = blockIdx.x * 8 + wave;
  const float* xp = XS + (size_t)t * C_DIM;
  float v[6], sum = 0.f, sq = 0.f;
#pragma unroll
  for (int i = 0; i < 6; ++i) {
    v[i] = xp[lane + 32 * i];
    sum += v[i]; sq += v[i] * v[i];
  }
#pragma unroll
  for (int m = 16; m >= 1; m >>= 1) {
    sum += __shfl_xor(sum, m, 32);
    sq  += __shfl_xor(sq, m, 32);
  }
  const float mu = sum * (1.f / C_DIM);
  const float rstd = rsqrtf(sq * (1.f / C_DIM) - mu * mu + 1e-5f);
  _Float16* op = H2 + (size_t)t * C_DIM;
#pragma unroll
  for (int i = 0; i < 6; ++i) {
    int ch = lane + 32 * i;
    op[ch] = (_Float16)((v[i] - mu) * rstd * g[ch] + be[ch]);
  }
}

// ---------------------------------------------------------------------------
// Generic WMMA f16 GEMM: C[M, NC] = A[M, K] * Bp + epilogue
// Block = 256 threads (8 waves). Block tile 256x64, each wave = 32x64
// (2 row groups x 4 col frags = 8 accumulators). Each B fragment feeds two
// WMMAs -> 1.5 b128 loads per WMMA. Epilogue uses a per-row context so
// divisions / coordinate math run once per row, not once per element.
// ---------------------------------------------------------------------------
template <int K, int NC, typename Epi>
__global__ __launch_bounds__(256) void gemm_wmma(const _Float16* __restrict__ A,
                                                 const _Float16* __restrict__ Bp,
                                                 Epi epi) {
  const int lane = threadIdx.x & 31;
  const int wave = threadIdx.x >> 5;
  const int halfId = lane >> 4;
  const int l16 = lane & 15;
  const int mBase = blockIdx.x * 256 + wave * 32;
  const int nBase = blockIdx.y * 64;
  const _Float16* arow0 = A + (size_t)(mBase + l16) * K;
  const _Float16* arow1 = arow0 + (size_t)16 * K;
  v8f acc0[4] = {v8f{}, v8f{}, v8f{}, v8f{}};
  v8f acc1[4] = {v8f{}, v8f{}, v8f{}, v8f{}};
  for (int k0 = 0; k0 < K; k0 += 32) {
    // A fragment (16x32 f16): lane row = l16, k = halfId*16 + e (contiguous)
    v16h a0 = mk16(*(const v8h*)(arow0 + k0 + halfId * 16),
                   *(const v8h*)(arow0 + k0 + halfId * 16 + 8));
    v16h a1 = mk16(*(const v8h*)(arow1 + k0 + halfId * 16),
                   *(const v8h*)(arow1 + k0 + halfId * 16 + 8));
    const _Float16* bbase =
        Bp + ((size_t)(k0 >> 5) * (NC / 16) + (nBase >> 4)) * 512 + lane * 16;
#pragma unroll
    for (int ni = 0; ni < 4; ++ni) {
      const _Float16* bp = bbase + ni * 512;
      v16h bf = mk16(*(const v8h*)bp, *(const v8h*)(bp + 8));
      acc0[ni] = wmma_f16(a0, bf, acc0[ni]);
      acc1[ni] = wmma_f16(a1, bf, acc1[ni]);
    }
  }
  // D layout: lane 0-15 -> M=j, lane 16-31 -> M=8+j; N = l16
#pragma unroll
  for (int j = 0; j < 8; ++j) {
    const int row0 = mBase + halfId * 8 + j;
    const auto c0 = epi.row(row0);
#pragma unroll
    for (int ni = 0; ni < 4; ++ni)
      epi.store(c0, nBase + ni * 16 + l16, acc0[ni][j]);
    const auto c1 = epi.row(row0 + 16);
#pragma unroll
    for (int ni = 0; ni < 4; ++ni)
      epi.store(c1, nBase + ni * 16 + l16, acc1[ni][j]);
  }
}

// ---------------------------------------------------------------------------
// Epilogue functors (two-phase: per-row context, then per-column store)
// ---------------------------------------------------------------------------
struct EpiQKV {  // torch reshape quirk: col = d*18 + h*3 + s
  const float* bias;
  _Float16 *Q, *Kb, *Vt;
  struct Ctx { size_t qk; size_t vt; };
  __device__ Ctx row(int r) const {
    const int b_ = r / 49, n = r % 49;
    const size_t base = (size_t)b_ * NHEAD * 2048;
    return Ctx{base + (size_t)n * DH, base + (size_t)n};
  }
  __device__ void store(const Ctx& c, int col, float v) const {
    v += bias[col];
    const int s = col % 3, h = (col / 3) % NHEAD, d = col / (3 * NHEAD);
    if (s == 0)
      Q[c.qk + (size_t)h * 2048 + d] = (_Float16)(v * 0.17677669529663687f);
    else if (s == 1)
      Kb[c.qk + (size_t)h * 2048 + d] = (_Float16)v;
    else
      Vt[c.vt + (size_t)h * 2048 + (size_t)d * 64] = (_Float16)v;  // V^T, N padded to 64
  }
};

struct EpiProj {  // + bias, window reverse, roll(+3,+3), + shortcut(x NCHW) -> XS NHWC f32
  const float* bias;
  const float* xin;
  float* XS;
  struct Ctx { size_t src; size_t dst; };
  __device__ Ctx row(int rw) const {
    const int b_ = rw / 49, n = rw % 49;
    const int b = b_ >> 6, wi = b_ & 63;
    const int r = ((wi >> 3) * 7 + n / 7 + 3) % 56;
    const int c = ((wi & 7) * 7 + n % 7 + 3) % 56;
    const size_t pix = (size_t)r * 56 + c;
    return Ctx{(size_t)b * C_DIM * 3136 + pix, ((size_t)b * 3136 + pix) * C_DIM};
  }
  __device__ void store(const Ctx& c, int col, float v) const {
    XS[c.dst + col] = xin[c.src + (size_t)col * 3136] + v + bias[col];
  }
};

struct EpiFC1 {  // + bias, exact GELU, -> f16
  const float* bias;
  _Float16* FF;
  struct Ctx { size_t o; };
  __device__ Ctx row(int r) const { return Ctx{(size_t)r * HN_DIM}; }
  __device__ void store(const Ctx& c, int col, float v) const {
    v += bias[col];
    FF[c.o + col] = (_Float16)(0.5f * v * (1.0f + erff(v * 0.70710678118654752f)));
  }
};

struct EpiFC2 {  // + bias + residual(XS), NHWC -> NCHW f32 output
  const float* bias;
  const float* XS;
  float* out;
  struct Ctx { size_t xs; size_t o; };
  __device__ Ctx row(int r) const {
    const int b = r / 3136, p = r % 3136;
    return Ctx{(size_t)r * C_DIM, (size_t)b * C_DIM * 3136 + p};
  }
  __device__ void store(const Ctx& c, int col, float v) const {
    out[c.o + (size_t)col * 3136] = XS[c.xs + col] + v + bias[col];
  }
};

// ---------------------------------------------------------------------------
// Attention: one block per (window, head). 128 threads = 4 waves; wave w owns
// query rows 16w..16w+15 (N padded 49->64). S = Q K^T (one WMMA per 16x16 tile,
// K-dim=32), + rel-pos bias + shift mask, register softmax (shfl within the
// 16-lane half-wave groups of the D layout), P staged via LDS as f16 A-frags,
// O = P V via V^T fragments (K-dim 64 = 2 steps).
// ---------------------------------------------------------------------------
__global__ __launch_bounds__(128) void attn_kernel(const _Float16* __restrict__ Q,
                                                   const _Float16* __restrict__ Km,
                                                   const _Float16* __restrict__ Vt,
                                                   const float* __restrict__ rpb,
                                                   _Float16* __restrict__ O) {
  const int bh = blockIdx.x;
  const int h = bh % NHEAD;
  const int b_ = bh / NHEAD;
  const int wi = b_ & 63;
  const int wy = wi >> 3, wx = wi & 7;
  const int lane = threadIdx.x & 31, wave = threadIdx.x >> 5;
  const int halfId = lane >> 4, l16 = lane & 15;

  const _Float16* Qb = Q  + (size_t)bh * 2048;
  const _Float16* Kb = Km + (size_t)bh * 2048;
  const _Float16* Vb = Vt + (size_t)bh * 2048;

  __shared__ _Float16 Pl[64 * 64];

  // ---- S = Q K^T : A = Q rows, B = K rows (K^T fragments are contiguous rows)
  const int rowQ = wave * 16 + l16;
  v16h af = mk16(*(const v8h*)(Qb + rowQ * DH + halfId * 16),
                 *(const v8h*)(Qb + rowQ * DH + halfId * 16 + 8));
  v8f s[4];
#pragma unroll
  for (int ni = 0; ni < 4; ++ni) {
    const _Float16* kp = Kb + (ni * 16 + l16) * DH + halfId * 16;
    v16h bf = mk16(*(const v8h*)kp, *(const v8h*)(kp + 8));
    v8f z{};
    s[ni] = wmma_f16(af, bf, z);
  }

  // ---- bias + mask + softmax (rows i = wave*16 + halfId*8 + j)
#pragma unroll
  for (int j = 0; j < 8; ++j) {
    const int i = wave * 16 + halfId * 8 + j;
    int regI = 0, yi = 0, xi = 0;
    if (i < 49) {
      yi = i / 7; xi = i % 7;
      const int ri = wy * 7 + yi, ci = wx * 7 + xi;
      regI = (ri < 49 ? 0 : (ri < 53 ? 1 : 2)) * 3 + (ci < 49 ? 0 : (ci < 53 ? 1 : 2));
    }
#pragma unroll
    for (int ni = 0; ni < 4; ++ni) {
      const int jc = ni * 16 + l16;
      float val = s[ni][j];
      if (jc >= 49) {
        val = -1e30f;                      // padded key columns -> softmax zero
      } else if (i < 49) {
        const int yj = jc / 7, xj = jc % 7;
        val += rpb[((yi - yj + 6) * 13 + (xi - xj + 6)) * NHEAD + h];
        const int rj = wy * 7 + yj, cj = wx * 7 + xj;
        const int regJ = (rj < 49 ? 0 : (rj < 53 ? 1 : 2)) * 3 +
                         (cj < 49 ? 0 : (cj < 53 ? 1 : 2));
        if (regJ != regI) val -= 100.0f;
      }
      s[ni][j] = val;
    }
    float mx = fmaxf(fmaxf(s[0][j], s[1][j]), fmaxf(s[2][j], s[3][j]));
#pragma unroll
    for (int m = 8; m >= 1; m >>= 1) mx = fmaxf(mx, __shfl_xor(mx, m, 32));
    float e[4], sm = 0.f;
#pragma unroll
    for (int ni = 0; ni < 4; ++ni) { e[ni] = __expf(s[ni][j] - mx); sm += e[ni]; }
#pragma unroll
    for (int m = 8; m >= 1; m >>= 1) sm += __shfl_xor(sm, m, 32);
    const float inv = 1.0f / sm;
#pragma unroll
    for (int ni = 0; ni < 4; ++ni)
      Pl[(size_t)i * 64 + ni * 16 + l16] = (_Float16)(e[ni] * inv);
  }
  __syncthreads();

  // ---- O = P V : A = P (from LDS), B = V^T rows (contiguous), K-dim = 64
  v8f acc2[2] = {v8f{}, v8f{}};
#pragma unroll
  for (int ks = 0; ks < 2; ++ks) {
    const _Float16* pr = &Pl[(size_t)(wave * 16 + l16) * 64 + ks * 32 + halfId * 16];
    v16h pa = mk16(*(const v8h*)pr, *(const v8h*)(pr + 8));
#pragma unroll
    for (int ni = 0; ni < 2; ++ni) {
      const _Float16* vp = Vb + (ni * 16 + l16) * 64 + ks * 32 + halfId * 16;
      v16h vf = mk16(*(const v8h*)vp, *(const v8h*)(vp + 8));
      acc2[ni] = wmma_f16(pa, vf, acc2[ni]);
    }
  }
#pragma unroll
  for (int ni = 0; ni < 2; ++ni)
#pragma unroll
    for (int j = 0; j < 8; ++j) {
      const int i = wave * 16 + halfId * 8 + j;
      if (i < 49)
        O[((size_t)b_ * 49 + i) * C_DIM + h * DH + ni * 16 + l16] =
            (_Float16)acc2[ni][j];
    }
}

// ---------------------------------------------------------------------------
extern "C" void kernel_launch(void* const* d_in, const int* in_sizes, int n_in,
                              void* d_out, int out_size, void* d_ws, size_t ws_size,
                              hipStream_t stream) {
  (void)in_sizes; (void)n_in; (void)out_size; (void)ws_size;
  const float* x      = (const float*)d_in[0];
  const float* rpb    = (const float*)d_in[1];
  const float* w_qkv  = (const float*)d_in[2];
  const float* b_qkv  = (const float*)d_in[3];
  const float* w_proj = (const float*)d_in[4];
  const float* b_proj = (const float*)d_in[5];
  const float* g0     = (const float*)d_in[6];
  const float* be0    = (const float*)d_in[7];
  const float* g1     = (const float*)d_in[8];
  const float* be1    = (const float*)d_in[9];
  const float* w_fc1  = (const float*)d_in[10];
  const float* b_fc1  = (const float*)d_in[11];
  const float* w_fc2  = (const float*)d_in[12];
  const float* b_fc2  = (const float*)d_in[13];
  float* out = (float*)d_out;

  char* ws = (char*)d_ws;
  _Float16* Xwin  = (_Float16*)(ws + XWIN_OFF);
  _Float16* Qb    = (_Float16*)(ws + Q_OFF);
  _Float16* Kbuf  = (_Float16*)(ws + K_OFF);
  _Float16* Vt    = (_Float16*)(ws + VT_OFF);
  _Float16* Obuf  = (_Float16*)(ws + OBUF_OFF);
  _Float16* H2    = (_Float16*)(ws + H2_OFF);
  float*    XS    = (float*)(ws + XS_OFF);
  _Float16* PBqkv = (_Float16*)(ws + PBQKV_OFF);
  _Float16* PBprj = (_Float16*)(ws + PBPROJ_OFF);
  _Float16* PBf1  = (_Float16*)(ws + PBFC1_OFF);
  _Float16* PBf2  = (_Float16*)(ws + PBFC2_OFF);
  _Float16* FF    = (_Float16*)(ws + FF_OFF);   // aliases Xwin/Q/K (dead by FC1)

  // 1) pack weights into WMMA B-fragment layout
  pack_w<<<(576 * 192 + 255) / 256, 256, 0, stream>>>(w_qkv,  PBqkv, 192, 576);
  pack_w<<<(192 * 192 + 255) / 256, 256, 0, stream>>>(w_proj, PBprj, 192, 192);
  pack_w<<<(768 * 192 + 255) / 256, 256, 0, stream>>>(w_fc1,  PBf1,  192, 768);
  pack_w<<<(192 * 768 + 255) / 256, 256, 0, stream>>>(w_fc2,  PBf2,  768, 192);

  // 2) LN0 + roll + window partition -> Xwin f16
  ln0_kernel<<<M_TOK / 8, 256, 0, stream>>>(x, g0, be0, Xwin);

  // 3) QKV GEMM (+ permuted scatter into Q, K, V^T; q pre-scaled)
  gemm_wmma<192, 576, EpiQKV><<<dim3(M_TOK / 256, 576 / 64), 256, 0, stream>>>(
      Xwin, PBqkv, EpiQKV{b_qkv, Qb, Kbuf, Vt});

  // 4) windowed attention (bias + shift mask + softmax + P*V)
  attn_kernel<<<4096 * NHEAD, 128, 0, stream>>>(Qb, Kbuf, Vt, rpb, Obuf);

  // 5) proj GEMM + window reverse + roll + residual -> XS (NHWC f32)
  gemm_wmma<192, 192, EpiProj><<<dim3(M_TOK / 256, 192 / 64), 256, 0, stream>>>(
      Obuf, PBprj, EpiProj{b_proj, x, XS});

  // 6) LN1 -> H2 f16
  ln1_kernel<<<M_TOK / 8, 256, 0, stream>>>(XS, g1, be1, H2);

  // 7) FC1 GEMM + GELU -> FF f16
  gemm_wmma<192, 768, EpiFC1><<<dim3(M_TOK / 256, 768 / 64), 256, 0, stream>>>(
      H2, PBf1, EpiFC1{b_fc1, FF});

  // 8) FC2 GEMM + residual + NHWC->NCHW -> d_out
  gemm_wmma<768, 192, EpiFC2><<<dim3(M_TOK / 256, 192 / 64), 256, 0, stream>>>(
      FF, PBf2, EpiFC2{b_fc2, XS, out});
}
